// EncoderLayer_12180527251813
// MI455X (gfx1250) — matrix operand order repaired and checked by reading, not compiled, not run on hardware
//
#include <hip/hip_runtime.h>

// ---------------------------------------------------------------------------
// Transformer encoder layer for MI455X (gfx1250, wave32, WMMA).
// GEMMs: v_wmma_f32_16x16x32_f16 with double-buffered LDS tiles filled by
// GLOBAL_LOAD_ASYNC_TO_LDS_B128 (ASYNCcnt-tracked DMA, bypasses VGPRs).
// ---------------------------------------------------------------------------

typedef __attribute__((ext_vector_type(16))) _Float16 v16h;
typedef __attribute__((ext_vector_type(8)))  _Float16 v8h;
typedef __attribute__((ext_vector_type(8)))  float    v8f;

#define B_    8
#define S_    1024
#define D_    1024
#define H_    16
#define DH_   64
#define FFN_  4096
#define M_    (B_ * S_)

static __device__ __forceinline__ v16h cat16(v8h lo, v8h hi) {
  return __builtin_shufflevector(lo, hi, 0,1,2,3,4,5,6,7,8,9,10,11,12,13,14,15);
}
static __device__ __forceinline__ v8f wmma16(v16h a, v16h b, v8f c) {
  // D = A(16x32 f16) * B(32x16 f16) + C(16x16 f32)
  return __builtin_amdgcn_wmma_f32_16x16x32_f16(false, a, false, b, (short)0, c,
                                                false, false);
}

// LDS byte offset of a __shared__ object: the LDS aperture is 4GB-aligned and
// FLAT addressing truncates to addr[31:0], so the generic pointer's low 32
// bits are exactly the hardware LDS address.
static __device__ __forceinline__ unsigned lds_off(const void* p) {
  return (unsigned)(unsigned long long)p;
}

// Async DMA: copy 32 contiguous bytes global -> LDS (two b128 ops sharing the
// same base operands, second uses INST_OFFSET which is added to both sides).
static __device__ __forceinline__ void async_copy32(unsigned lds,
                                                    const void* gptr) {
  asm volatile(
      "global_load_async_to_lds_b128 %0, %1, off\n\t"
      "global_load_async_to_lds_b128 %0, %1, off offset:16"
      :
      : "v"(lds), "v"((unsigned long long)gptr)
      : "memory");
}

static __device__ __forceinline__ void wait_async0() {
#if __has_builtin(__builtin_amdgcn_s_wait_asynccnt)
  __builtin_amdgcn_s_wait_asynccnt(0);
#else
  asm volatile("s_wait_asynccnt 0x0" ::: "memory");
#endif
}

// --- elementwise f32 -> f16 convert --------------------------------------
__global__ __launch_bounds__(256) void cvt_f16_kernel(
    const float* __restrict__ in, _Float16* __restrict__ out, int n) {
  int i = blockIdx.x * 256 + threadIdx.x;
  if (i < n) out[i] = (_Float16)in[i];
}

// --- pack per-head weight [H, D, DH] -> BT layout [N=H*DH][K=D] f16 -------
__global__ __launch_bounds__(256) void pack_headw_kernel(
    const float* __restrict__ W, _Float16* __restrict__ out) {
  int i = blockIdx.x * 256 + threadIdx.x;   // over D_*D_ outputs
  if (i >= D_ * D_) return;
  int n = i / D_;            // output row = h*DH + e
  int kd = i - n * D_;       // column  = d
  int h = n >> 6, e = n & 63;
  out[i] = (_Float16)W[((size_t)h * D_ + kd) * DH_ + e];
}

// --- transpose V: [b,s,h*DH+e] f16 -> VT [ (b*H+h)*DH+e ][ t ] f16 --------
__global__ __launch_bounds__(256) void transpose_v_kernel(
    const _Float16* __restrict__ v, _Float16* __restrict__ vt) {
  int i = blockIdx.x * 256 + threadIdx.x;   // over B_*H_*DH_*S_
  if (i >= B_ * H_ * DH_ * S_) return;
  int t = i & (S_ - 1);
  int rest = i >> 10;            // (b*H+h)*DH + e
  int e = rest & (DH_ - 1);
  int bh = rest >> 6;
  int h = bh & (H_ - 1);
  int b = bh >> 4;
  vt[i] = v[((size_t)(b * S_ + t)) * D_ + h * DH_ + e];
}

// --- tiled WMMA GEMM: C[M,N] = A[M,K] @ B[K,N], B given transposed [N][K] --
// 128 threads = 4 waves; workgroup tile 64x64; each wave 32x32 (2x2 frags).
// Double-buffered LDS tiles filled by async global->LDS DMA; the DMA for
// tile k+1 overlaps the ds_load+WMMA work on tile k.
__global__ __launch_bounds__(128) void gemm_wmma_kernel(
    const _Float16* __restrict__ A, const _Float16* __restrict__ BT,
    const float* __restrict__ bias, float* __restrict__ Cf32,
    _Float16* __restrict__ Cf16, int Mdim, int Ndim, int Kdim, int relu) {
  __shared__ _Float16 As[2][64][40];  // [buf][m][k], pad 8 halves (16B rows)
  __shared__ _Float16 Bs[2][64][40];  // [buf][n][k]
  const int tid = threadIdx.x;
  const int wave = tid >> 5, lane = tid & 31;
  const int wm = (wave >> 1) * 32;     // wave row offset in tile
  const int wn = (wave & 1) * 32;      // wave col offset in tile
  const int m0 = blockIdx.y * 64;
  const int n0 = blockIdx.x * 64;
  const int mfrag = lane & 15, khalf = lane >> 4;

  v8f acc[2][2] = {};

  const int lr = tid >> 1;             // 0..63: tile row (A) / tile n-row (B)
  const int lc = (tid & 1) * 16;       // 0 or 16: which 16-half chunk

  const _Float16* ga = A + (size_t)(m0 + lr) * Kdim + lc;   // this lane's rows
  const _Float16* gb = BT + (size_t)(n0 + lr) * Kdim + lc;
  const unsigned la0 = lds_off(&As[0][lr][lc]);
  const unsigned lb0 = lds_off(&Bs[0][lr][lc]);
  const unsigned la1 = lds_off(&As[1][lr][lc]);
  const unsigned lb1 = lds_off(&Bs[1][lr][lc]);

  // prologue: DMA tile 0
  async_copy32(la0, ga);
  async_copy32(lb0, gb);
  wait_async0();
  __syncthreads();

  const int nk = Kdim >> 5;
  for (int kt = 0; kt < nk; ++kt) {
    const int buf = kt & 1;
    // issue DMA for next tile into the other buffer (overlaps WMMA below)
    if (kt + 1 < nk) {
      async_copy32(buf ? la0 : la1, ga + (kt + 1) * 32);
      async_copy32(buf ? lb0 : lb1, gb + (kt + 1) * 32);
    }

    // A fragments: lane half selects K groups of 8 (ISA 16-bit A layout)
    v16h af[2], bf[2];
#pragma unroll
    for (int i = 0; i < 2; ++i) {
      int row = wm + i * 16 + mfrag;
      v8h lo = *(const v8h*)&As[buf][row][khalf * 8];       // K = kh*8 + 0..7
      v8h hi = *(const v8h*)&As[buf][row][16 + khalf * 8];  // K = 16+kh*8+0..7
      af[i] = cat16(lo, hi);
    }
    // B fragments: lane half selects K groups of 8 (same order as A)
#pragma unroll
    for (int j = 0; j < 2; ++j) {
      int col = wn + j * 16 + mfrag;                        // N index
      v8h lo = *(const v8h*)&Bs[buf][col][khalf * 8];       // K = kh*8 + 0..7
      v8h hi = *(const v8h*)&Bs[buf][col][16 + khalf * 8];  // K = 16+kh*8+0..7
      bf[j] = cat16(lo, hi);
    }
#pragma unroll
    for (int i = 0; i < 2; ++i)
#pragma unroll
      for (int j = 0; j < 2; ++j)
        acc[i][j] = wmma16(af[i], bf[j], acc[i][j]);

    // drain this wave's DMAs, then make them visible to all waves
    wait_async0();
    __syncthreads();
  }

  // store: C/D layout: element r -> M = r + khalf*8, N = mfrag
#pragma unroll
  for (int j = 0; j < 2; ++j) {
    int col = n0 + wn + j * 16 + mfrag;
    float bv = bias ? bias[col] : 0.0f;
#pragma unroll
    for (int i = 0; i < 2; ++i) {
#pragma unroll
      for (int r = 0; r < 8; ++r) {
        int row = m0 + wm + i * 16 + khalf * 8 + r;
        float v = acc[i][j][r] + bv;
        if (relu && v < 0.0f) v = 0.0f;
        size_t idx = (size_t)row * Ndim + col;
        if (Cf32) Cf32[idx] = v;
        if (Cf16) Cf16[idx] = (_Float16)v;
      }
    }
  }
}

// --- flash-style attention: one wave per (b, h, 16-row query block) -------
__global__ __launch_bounds__(32) void attn_kernel(
    const _Float16* __restrict__ q, const _Float16* __restrict__ k,
    const _Float16* __restrict__ vt, const int* __restrict__ real_lengths,
    _Float16* __restrict__ heads) {
  __shared__ float    sc[16][32];    // score tile (f32)
  __shared__ _Float16 pl[16][32];    // P tile (f16)
  __shared__ float rmax[16], rsum[16], ralpha[16];

  const int lane = threadIdx.x;
  const int s0 = blockIdx.x * 16;
  const int h = blockIdx.y, b = blockIdx.z;
  const int len = real_lengths[b];
  const int mfrag = lane & 15, khalf = lane >> 4;

  // Q block 16x64 as two A fragments (K = DH = 64)
  const _Float16* qrow = q + ((size_t)(b * S_ + s0 + mfrag)) * D_ + h * DH_;
  v16h aq[2];
#pragma unroll
  for (int f = 0; f < 2; ++f) {
    v8h lo = *(const v8h*)(qrow + f * 32 + khalf * 8);
    v8h hi = *(const v8h*)(qrow + f * 32 + 16 + khalf * 8);
    aq[f] = cat16(lo, hi);
  }
  if (lane < 16) { rmax[lane] = -3.0e38f; rsum[lane] = 0.0f; }
  __syncthreads();

  v8f o[4] = {};
  for (int t0 = 0; t0 < S_ && t0 < len; t0 += 32) {
    // scores: 16 rows x 32 keys = two 16x16 tiles, K=64 -> 2 WMMA each
    v8f st[2] = {};
#pragma unroll
    for (int f = 0; f < 2; ++f) {
#pragma unroll
      for (int nt = 0; nt < 2; ++nt) {
        const _Float16* krow =
            k + ((size_t)(b * S_ + t0 + nt * 16 + mfrag)) * D_ + h * DH_ + f * 32;
        v8h lo = *(const v8h*)(krow + khalf * 8);
        v8h hi = *(const v8h*)(krow + 16 + khalf * 8);
        st[nt] = wmma16(aq[f], cat16(lo, hi), st[nt]);
      }
    }
    // scale by 1/sqrt(DH), apply key-padding mask, spill tile to LDS
#pragma unroll
    for (int nt = 0; nt < 2; ++nt) {
      int t = t0 + nt * 16 + mfrag;
#pragma unroll
      for (int r = 0; r < 8; ++r)
        sc[khalf * 8 + r][nt * 16 + mfrag] =
            (t < len) ? st[nt][r] * 0.125f : -1.0e9f;
    }
    __syncthreads();

    // online softmax: lane m owns row m
    if (lane < 16) {
      float mo = rmax[lane], mn = mo;
      for (int c = 0; c < 32; ++c) mn = fmaxf(mn, sc[lane][c]);
      float alpha = __expf(mo - mn);
      float sum = rsum[lane] * alpha;
      for (int c = 0; c < 32; ++c) {
        float p = __expf(sc[lane][c] - mn);
        sum += p;
        pl[lane][c] = (_Float16)p;
      }
      rmax[lane] = mn; rsum[lane] = sum; ralpha[lane] = alpha;
    }
    __syncthreads();

    float al[8];
#pragma unroll
    for (int r = 0; r < 8; ++r) al[r] = ralpha[khalf * 8 + r];

    // P (16x32 f16) as A fragment
    v8h plo = *(const v8h*)&pl[mfrag][khalf * 8];
    v8h phi = *(const v8h*)&pl[mfrag][16 + khalf * 8];
    v16h pa = cat16(plo, phi);

    // O = alpha*O + P @ V_block  (V pre-transposed: [bh*64+e][t])
#pragma unroll
    for (int e = 0; e < 4; ++e) {
#pragma unroll
      for (int r = 0; r < 8; ++r) o[e][r] *= al[r];
      const _Float16* vrow =
          vt + ((size_t)((b * H_ + h) * DH_ + e * 16 + mfrag)) * S_ + t0 +
          khalf * 8;
      v8h vlo = *(const v8h*)vrow;
      v8h vhi = *(const v8h*)(vrow + 16);
      o[e] = wmma16(pa, cat16(vlo, vhi), o[e]);
    }
    __syncthreads();
  }

  float inv[8];
#pragma unroll
  for (int r = 0; r < 8; ++r) inv[r] = 1.0f / rsum[khalf * 8 + r];
#pragma unroll
  for (int e = 0; e < 4; ++e)
#pragma unroll
    for (int r = 0; r < 8; ++r) {
      int row = s0 + khalf * 8 + r;
      heads[((size_t)(b * S_ + row)) * D_ + h * DH_ + e * 16 + mfrag] =
          (_Float16)(o[e][r] * inv[r]);
    }
}

// --- residual add + LayerNorm over D=1024 (one block per row) ------------
__global__ __launch_bounds__(256) void add_ln_kernel(
    const float* __restrict__ xa, const float* __restrict__ xb,
    float* __restrict__ of32, _Float16* __restrict__ of16) {
  __shared__ float sm[256];
  const int row = blockIdx.x, tid = threadIdx.x;
  const float* pa = xa + (size_t)row * D_;
  const float* pb = xb + (size_t)row * D_;
  float v[4];
  float s = 0.0f;
#pragma unroll
  for (int i = 0; i < 4; ++i) {
    v[i] = pa[tid + 256 * i] + pb[tid + 256 * i];
    s += v[i];
  }
  sm[tid] = s; __syncthreads();
  for (int st = 128; st > 0; st >>= 1) {
    if (tid < st) sm[tid] += sm[tid + st];
    __syncthreads();
  }
  float mean = sm[0] * (1.0f / D_);
  __syncthreads();
  float vs = 0.0f;
#pragma unroll
  for (int i = 0; i < 4; ++i) { float d = v[i] - mean; vs += d * d; }
  sm[tid] = vs; __syncthreads();
  for (int st = 128; st > 0; st >>= 1) {
    if (tid < st) sm[tid] += sm[tid + st];
    __syncthreads();
  }
  float inv = rsqrtf(sm[0] * (1.0f / D_) + 1e-5f);
#pragma unroll
  for (int i = 0; i < 4; ++i) {
    float o = (v[i] - mean) * inv;
    size_t idx = (size_t)row * D_ + tid + 256 * i;
    of32[idx] = o;
    if (of16) of16[idx] = (_Float16)o;
  }
}

// ---------------------------------------------------------------------------
extern "C" void kernel_launch(void* const* d_in, const int* in_sizes, int n_in,
                              void* d_out, int out_size, void* d_ws,
                              size_t ws_size, hipStream_t stream) {
  const float* x  = (const float*)d_in[0];
  const int* rl   = (const int*)d_in[1];
  const float* Wq = (const float*)d_in[2];
  const float* Wk = (const float*)d_in[3];
  const float* Wv = (const float*)d_in[4];
  const float* Wo = (const float*)d_in[5];
  const float* W1 = (const float*)d_in[6];
  const float* b1 = (const float*)d_in[7];
  const float* W2 = (const float*)d_in[8];
  const float* b2 = (const float*)d_in[9];
  float* out = (float*)d_out;

  char* ws = (char*)d_ws;
  size_t off = 0;
  auto alloc = [&](size_t bytes) {
    void* p = ws + off;
    off += (bytes + 255) & ~(size_t)255;
    return p;
  };
  _Float16* xh    = (_Float16*)alloc((size_t)M_ * D_ * 2);
  _Float16* wqT   = (_Float16*)alloc((size_t)D_ * D_ * 2);
  _Float16* wkT   = (_Float16*)alloc((size_t)D_ * D_ * 2);
  _Float16* wvT   = (_Float16*)alloc((size_t)D_ * D_ * 2);
  _Float16* woT   = (_Float16*)alloc((size_t)D_ * D_ * 2);
  _Float16* w1T   = (_Float16*)alloc((size_t)FFN_ * D_ * 2);
  _Float16* w2T   = (_Float16*)alloc((size_t)D_ * FFN_ * 2);
  _Float16* qf    = (_Float16*)alloc((size_t)M_ * D_ * 2);
  _Float16* kf    = (_Float16*)alloc((size_t)M_ * D_ * 2);
  _Float16* vf    = (_Float16*)alloc((size_t)M_ * D_ * 2);
  _Float16* vtf   = (_Float16*)alloc((size_t)B_ * H_ * DH_ * S_ * 2);
  _Float16* headf = (_Float16*)alloc((size_t)M_ * D_ * 2);
  float*    attn  = (float*)alloc((size_t)M_ * D_ * 4);
  float*    x1    = (float*)alloc((size_t)M_ * D_ * 4);
  _Float16* x1h   = (_Float16*)alloc((size_t)M_ * D_ * 2);
  _Float16* h1    = (_Float16*)alloc((size_t)M_ * FFN_ * 2);
  float*    ffn   = (float*)alloc((size_t)M_ * D_ * 4);
  (void)ws_size; (void)in_sizes; (void)n_in; (void)out_size;

  const int nx = M_ * D_;
  // 1) convert activations + pack weights to f16 [N][K] ("BT") layouts
  cvt_f16_kernel<<<(nx + 255) / 256, 256, 0, stream>>>(x, xh, nx);
  pack_headw_kernel<<<(D_ * D_ + 255) / 256, 256, 0, stream>>>(Wq, wqT);
  pack_headw_kernel<<<(D_ * D_ + 255) / 256, 256, 0, stream>>>(Wk, wkT);
  pack_headw_kernel<<<(D_ * D_ + 255) / 256, 256, 0, stream>>>(Wv, wvT);
  cvt_f16_kernel<<<(D_ * D_ + 255) / 256, 256, 0, stream>>>(Wo, woT, D_ * D_);
  cvt_f16_kernel<<<(FFN_ * D_ + 255) / 256, 256, 0, stream>>>(W1, w1T, FFN_ * D_);
  cvt_f16_kernel<<<(D_ * FFN_ + 255) / 256, 256, 0, stream>>>(W2, w2T, D_ * FFN_);

  // 2) QKV projections (M=8192, K=1024, N=1024), f16 outputs
  dim3 blk(128);
  dim3 g1(D_ / 64, M_ / 64);
  gemm_wmma_kernel<<<g1, blk, 0, stream>>>(xh, wqT, nullptr, nullptr, qf,
                                           M_, D_, D_, 0);
  gemm_wmma_kernel<<<g1, blk, 0, stream>>>(xh, wkT, nullptr, nullptr, kf,
                                           M_, D_, D_, 0);
  gemm_wmma_kernel<<<g1, blk, 0, stream>>>(xh, wvT, nullptr, nullptr, vf,
                                           M_, D_, D_, 0);

  // 3) transpose V for the P@V WMMA B-operand
  const int nv = B_ * H_ * DH_ * S_;
  transpose_v_kernel<<<(nv + 255) / 256, 256, 0, stream>>>(vf, vtf);

  // 4) masked softmax attention, heads concat in f16
  attn_kernel<<<dim3(S_ / 16, H_, B_), dim3(32), 0, stream>>>(qf, kf, vtf, rl,
                                                              headf);

  // 5) output projection -> attn (f32)
  gemm_wmma_kernel<<<g1, blk, 0, stream>>>(headf, woT, nullptr, attn, nullptr,
                                           M_, D_, D_, 0);

  // 6) x1 = LN(x + attn), keep f32 and f16 copies
  add_ln_kernel<<<M_, 256, 0, stream>>>(x, attn, x1, x1h);

  // 7) FFN: relu(x1 @ W1^T + b1) -> h1 (f16), then @ W2^T + b2 -> ffn (f32)
  dim3 g2(FFN_ / 64, M_ / 64);
  gemm_wmma_kernel<<<g2, blk, 0, stream>>>(x1h, w1T, b1, nullptr, h1,
                                           M_, FFN_, D_, 1);
  gemm_wmma_kernel<<<g1, blk, 0, stream>>>(h1, w2T, b2, ffn, nullptr,
                                           M_, D_, FFN_, 0);

  // 8) out = LN(x1 + ffn)
  add_ln_kernel<<<M_, 256, 0, stream>>>(x1, ffn, out, nullptr);
}
